// PrototypicalNetwork_24842090840740
// MI455X (gfx1250) — compile-verified, matrix operand order checked by reading, not run on hardware
//
#include <hip/hip_runtime.h>

// ---------------------------------------------------------------------------
// Mahalanobis prototypical classifier for MI455X (gfx1250, wave32).
//  - All O(D^3)/O(N*D^2) math through v_wmma_f32_16x16x32_bf16.
//  - All GEMMs are NT (C = A*B^T) on row-major operands: every B operand is
//    symmetric (covariances, Newton-Schulz iterates, precision P), so no
//    transposed staging exists anywhere.
//  - K-tiles staged global->LDS by the Tensor Data Mover (tensor_load_to_lds,
//    TENSORcnt, double-buffered); TDM's pad feature emits the padded
//    bank-conflict-free LDS stride directly.
//  - Matrix inverse via Newton-Schulz (GEMM-only), X0 = I/trace(M).
// Workspace requirement: ~233 MB.
// ---------------------------------------------------------------------------

typedef unsigned short u16;
typedef unsigned int   u32;
typedef unsigned long long u64;

typedef __attribute__((ext_vector_type(16))) __bf16 v16bf;
typedef __attribute__((ext_vector_type(8)))  float  v8f;
typedef __attribute__((ext_vector_type(4)))  unsigned int v4u;
typedef __attribute__((ext_vector_type(8)))  int    v8i;
typedef __attribute__((ext_vector_type(4)))  int    v4i;

#define DIM   2048     // feature dim D
#define SSUP  2048     // support rows S
#define NQ    16384    // query rows N
#define NCLS  2

#define BM 128
#define BN 128
#define BK 32
#define LDT 40         // padded LDS row stride (bf16) = 20 dwords (conflict-free frag reads)
#define NS_ITERS 20

// ----- helpers --------------------------------------------------------------

__device__ __forceinline__ u16 f2bf(float f) {
  u32 u = __float_as_uint(f);
  u32 r = (u + 0x7fffu + ((u >> 16) & 1u)) >> 16;   // round-to-nearest-even
  return (u16)r;
}

// 16x32 bf16 fragment from a row-major padded LDS tile (A and B^T identical).
// lane&15 = row, lane>>4 = K-octet; VGPR v packs K-pairs per ISA layout.
__device__ __forceinline__ v16bf frag_from_lds(const u16* tile, int row, int lane) {
  union { v16bf v; u32 u[8]; } f;
  const u32* p = (const u32*)(tile + row * LDT);
  const int h4 = ((lane >> 4) & 1) * 4;
#pragma unroll
  for (int i = 0; i < 4; ++i) { f.u[i] = p[h4 + i]; f.u[i + 4] = p[h4 + 8 + i]; }
  return f.v;
}

// TDM: load a BMxBK (rows x K) bf16 tile, row stride ld (elements), into LDS
// at lds_addr with 16-dword data + 4-dword pad per row => LDT-element stride.
__device__ __forceinline__ void tdm_load_tile(u32 lds_addr, const u16* gptr,
                                              int kbound, int rowbound, int ld) {
  u64 ga = (u64)(uintptr_t)gptr;
  v4u g0 = { 1u,                                  // count=1 (valid descriptor)
             lds_addr,                            // LDS byte address
             (u32)ga,                             // global addr [31:0]
             (u32)((ga >> 32) & 0x01ffffffu) | 0x80000000u };  // addr[56:32] | type=2
  v8i g1 = { (int)((1u << 16) |                   // data_size = 2 bytes
                   (1u << 20) |                   // pad_enable
                   (3u << 22) |                   // pad_interval: 16 dwords
                   (3u << 25)),                   // pad_amount: 4 dwords
             (int)(((u32)kbound & 0xffffu) << 16),            // tensor_dim0 lo16
             (int)(((u32)kbound >> 16) |
                   (((u32)rowbound & 0xffffu) << 16)),        // dim0 hi | dim1 lo
             (int)(((u32)rowbound >> 16) | ((u32)BK << 16)),  // dim1 hi | tile_dim0
             (int)BM,                                         // tile_dim1 = 128 rows
             (int)ld,                                         // tensor_dim0_stride lo32
             0, 0 };
  v4i z4 = { 0, 0, 0, 0 };
  v8i z8 = { 0, 0, 0, 0, 0, 0, 0, 0 };
  __builtin_amdgcn_tensor_load_to_lds(g0, g1, z4, z4, z8, 0);
}

// ----- stats / packing ------------------------------------------------------

__global__ __launch_bounds__(256)
void k_count(const int* __restrict__ labels, float* __restrict__ counts) {
  __shared__ int red[256];
  int t = threadIdx.x, c0 = 0;
  for (int s = t; s < SSUP; s += 256) c0 += (labels[s] == 0) ? 1 : 0;
  red[t] = c0; __syncthreads();
  for (int o = 128; o > 0; o >>= 1) { if (t < o) red[t] += red[t + o]; __syncthreads(); }
  if (t == 0) { counts[0] = (float)red[0]; counts[1] = (float)(SSUP - red[0]); }
}

__global__ __launch_bounds__(256)
void k_colsum(const float* __restrict__ X, const int* __restrict__ labels,
              const float* __restrict__ counts,
              float* __restrict__ mean0, float* __restrict__ mean1,
              float* __restrict__ gmean) {
  __shared__ int lab[SSUP];
  int t = threadIdx.x;
  for (int s = t; s < SSUP; s += 256) lab[s] = labels[s];
  __syncthreads();
  int col = blockIdx.x * 256 + t;
  float s0 = 0.f, s1 = 0.f;
  for (int s = 0; s < SSUP; ++s) {
    float x = X[(size_t)s * DIM + col];
    if (lab[s] == 0) s0 += x; else s1 += x;
  }
  mean0[col] = s0 / counts[0];
  mean1[col] = s1 / counts[1];
  gmean[col] = (s0 + s1) * (1.0f / (float)SSUP);
}

// Transposed centered/masked bf16 packs: xtT[d][s] = x - gmean, xcT_c = masked.
__global__ __launch_bounds__(256)
void k_packT(const float* __restrict__ X, const int* __restrict__ labels,
             const float* __restrict__ mean0, const float* __restrict__ mean1,
             const float* __restrict__ gmean,
             u16* __restrict__ xtT, u16* __restrict__ xc0T, u16* __restrict__ xc1T) {
  __shared__ u16 sT[64 * 65], s0m[64 * 65], s1m[64 * 65];
  __shared__ int lab[64];
  int t = threadIdx.x;
  int s0 = blockIdx.x * 64, d0 = blockIdx.y * 64;
  if (t < 64) lab[t] = labels[s0 + t];
  __syncthreads();
#pragma unroll
  for (int r = 0; r < 16; ++r) {
    int sl = r * 4 + (t >> 6);
    int dl = t & 63;
    float x = X[(size_t)(s0 + sl) * DIM + d0 + dl];
    int l = lab[sl];
    sT [dl * 65 + sl] = f2bf(x - gmean[d0 + dl]);
    s0m[dl * 65 + sl] = f2bf(l == 0 ? (x - mean0[d0 + dl]) : 0.f);
    s1m[dl * 65 + sl] = f2bf(l == 1 ? (x - mean1[d0 + dl]) : 0.f);
  }
  __syncthreads();
#pragma unroll
  for (int r = 0; r < 16; ++r) {
    int dl = r * 4 + (t >> 6);
    int sl = t & 63;
    size_t o = (size_t)(d0 + dl) * SSUP + s0 + sl;
    xtT[o] = sT[dl * 65 + sl]; xc0T[o] = s0m[dl * 65 + sl]; xc1T[o] = s1m[dl * 65 + sl];
  }
}

// ----- NT GEMM core: C[m,n] = sum_k A[m,k]*B[n,k] (bf16 in, f32 accum) ------
// FUSED=false: plain store of C.
// FUSED=true : query epilogue, raw[m,cls] += sum_n C[m,n]*(Q[m,n]-2*mu[n]).

template <bool FUSED>
__global__ __launch_bounds__(256)
void k_gemm_nt(const u16* __restrict__ A, const u16* __restrict__ B,
               float* __restrict__ C, int K, int lda, int ldb, int ldc,
               int Mrows, int Nrows,
               const float* __restrict__ Qf, const float* __restrict__ mean,
               float* __restrict__ raw, int cls) {
  __shared__ __align__(16) u16 sA[2][BM * LDT];
  __shared__ __align__(16) u16 sB[2][BN * LDT];
  const int t = threadIdx.x;
  const int lane = t & 31, wid = t >> 5;
  const int wm = wid & 3, wn = wid >> 2;          // 4x2 wave grid
  const int bm0 = blockIdx.y * BM, bn0 = blockIdx.x * BN;
  const int NK = K / BK;

  v8f acc[2][4];
#pragma unroll
  for (int i = 0; i < 2; ++i)
#pragma unroll
    for (int j = 0; j < 4; ++j) {
      v8f z = {0.f, 0.f, 0.f, 0.f, 0.f, 0.f, 0.f, 0.f};
      acc[i][j] = z;
    }

  const u32 ldsA0 = (u32)(uintptr_t)(void*)&sA[0][0];
  const u32 ldsA1 = (u32)(uintptr_t)(void*)&sA[1][0];
  const u32 ldsB0 = (u32)(uintptr_t)(void*)&sB[0][0];
  const u32 ldsB1 = (u32)(uintptr_t)(void*)&sB[1][0];

  if (wid == 0) {                                 // prologue: tile 0 in flight
    tdm_load_tile(ldsA0, A + (size_t)bm0 * lda, K, Mrows, lda);
    tdm_load_tile(ldsB0, B + (size_t)bn0 * ldb, K, Nrows, ldb);
  }

  for (int kt = 0; kt < NK; ++kt) {
    const int cur = kt & 1;
    if (wid == 0) {
      if (kt + 1 < NK) {                          // issue next tile, retire current
        const int k1 = (kt + 1) * BK;
        tdm_load_tile(cur ? ldsA0 : ldsA1, A + (size_t)bm0 * lda + k1, K, Mrows, lda);
        tdm_load_tile(cur ? ldsB0 : ldsB1, B + (size_t)bn0 * ldb + k1, K, Nrows, ldb);
        __builtin_amdgcn_s_wait_tensorcnt(2);
      } else {
        __builtin_amdgcn_s_wait_tensorcnt(0);
      }
    }
    __syncthreads();

    v16bf af[2], bfr[4];
#pragma unroll
    for (int tm = 0; tm < 2; ++tm)
      af[tm] = frag_from_lds(sA[cur], wm * 32 + tm * 16 + (lane & 15), lane);
#pragma unroll
    for (int tn = 0; tn < 4; ++tn)
      bfr[tn] = frag_from_lds(sB[cur], wn * 64 + tn * 16 + (lane & 15), lane);
#pragma unroll
    for (int tm = 0; tm < 2; ++tm)
#pragma unroll
      for (int tn = 0; tn < 4; ++tn)
        acc[tm][tn] = __builtin_amdgcn_wmma_f32_16x16x32_bf16(
            false, af[tm], false, bfr[tn], (short)0, acc[tm][tn], false, false);
    __syncthreads();                              // all reads done before next TDM write
  }

  const int r15 = lane & 15, h = lane >> 4;       // C layout: VGPR r -> M=r+8h, lane -> N
  if (!FUSED) {
#pragma unroll
    for (int tm = 0; tm < 2; ++tm)
#pragma unroll
      for (int tn = 0; tn < 4; ++tn) {
        int n = bn0 + wn * 64 + tn * 16 + r15;
#pragma unroll
        for (int r = 0; r < 8; ++r) {
          int m = bm0 + wm * 32 + tm * 16 + h * 8 + r;
          C[(size_t)m * ldc + n] = acc[tm][tn][r];
        }
      }
  } else {
#pragma unroll
    for (int tm = 0; tm < 2; ++tm) {
#pragma unroll
      for (int r = 0; r < 8; ++r) {
        int mg = bm0 + wm * 32 + tm * 16 + h * 8 + r;
        float partial = 0.f;
#pragma unroll
        for (int tn = 0; tn < 4; ++tn) {
          int ng = bn0 + wn * 64 + tn * 16 + r15;
          float w = Qf[(size_t)mg * DIM + ng] - 2.0f * mean[ng];
          partial += acc[tm][tn][r] * w;
        }
        for (int off = 1; off < 16; off <<= 1)    // reduce across the 16 N-lanes
          partial += __shfl_xor(partial, off, 32);
        if (r15 == 0) atomicAdd(&raw[mg * NCLS + cls], partial);
      }
    }
  }
}

// ----- elementwise / reductions ---------------------------------------------

__global__ __launch_bounds__(256)
void k_combine(float* __restrict__ G, const float* __restrict__ T,
               const float* __restrict__ counts, float* __restrict__ tr, int cls) {
  size_t idx = (size_t)blockIdx.x * 256 + threadIdx.x;
  int i = (int)(idx / DIM), j = (int)(idx % DIM);
  float n = counts[cls];
  float lam = fminf(n / (n + 1.0f), 0.1f);
  float a = lam / (n - 1.0f);
  float b = (1.0f - lam) / (float)(SSUP - 1);
  float v = a * G[idx] + b * T[idx] + ((i == j) ? 0.1f : 0.0f);
  G[idx] = v;
  if (i == j) atomicAdd(&tr[cls], v);
}

__global__ __launch_bounds__(256)
void k_cvt(const float* __restrict__ src, u16* __restrict__ dst) {
  size_t idx = (size_t)blockIdx.x * 256 + threadIdx.x;
  dst[idx] = f2bf(src[idx]);
}

__global__ __launch_bounds__(256)
void k_ns_init(const float* __restrict__ tr, float* __restrict__ X,
               u16* __restrict__ Xb, int cls) {
  size_t idx = (size_t)blockIdx.x * 256 + threadIdx.x;
  int i = (int)(idx / DIM), j = (int)(idx % DIM);
  float v = (i == j) ? (1.0f / tr[cls]) : 0.0f;
  X[idx] = v; Xb[idx] = f2bf(v);
}

__global__ __launch_bounds__(256)
void k_update(float* __restrict__ X, const float* __restrict__ U,
              u16* __restrict__ Xb) {                // X = 2X - U
  size_t idx = (size_t)blockIdx.x * 256 + threadIdx.x;
  float v = 2.0f * X[idx] - U[idx];
  X[idx] = v; Xb[idx] = f2bf(v);
}

// s_c = mu^T P mu
__global__ __launch_bounds__(256)
void k_sc(const float* __restrict__ P, const float* __restrict__ mean,
          float* __restrict__ sc, int cls) {
  __shared__ float red[256];
  int d = blockIdx.x, t = threadIdx.x;
  float a = 0.f;
  for (int e = t; e < DIM; e += 256) a += P[(size_t)d * DIM + e] * mean[e];
  red[t] = a; __syncthreads();
  for (int o = 128; o > 0; o >>= 1) { if (t < o) red[t] += red[t + o]; __syncthreads(); }
  if (t == 0) atomicAdd(&sc[cls], red[0] * mean[d]);
}

__global__ __launch_bounds__(256)
void k_final(const float* __restrict__ raw, const float* __restrict__ sc,
             float* __restrict__ out) {
  int idx = blockIdx.x * 256 + threadIdx.x;
  int c = idx & 1;
  out[idx] = -(raw[idx] + sc[c]);
}

// ----- host-side orchestration ----------------------------------------------

extern "C" void kernel_launch(void* const* d_in, const int* in_sizes, int n_in,
                              void* d_out, int out_size, void* d_ws, size_t ws_size,
                              hipStream_t stream) {
  (void)in_sizes; (void)n_in; (void)out_size; (void)ws_size;
  const float* Q      = (const float*)d_in[0];
  const float* X      = (const float*)d_in[1];
  const int*   labels = (const int*)d_in[2];
  float* out = (float*)d_out;

  char* ws = (char*)d_ws;
  const size_t MB = 1ull << 20;
  float* counts = (float*)(ws + 0);
  float* tr     = (float*)(ws + 64);
  float* sc     = (float*)(ws + 128);
  float* mean0  = (float*)(ws + 1024);
  float* mean1  = (float*)(ws + 1024 + 8192);
  float* gmean  = (float*)(ws + 1024 + 16384);
  char* big = ws + 32768;
  u16*   xtT   = (u16*)(big);                     // [D,S] transposed packs
  u16*   xc0T  = (u16*)(big + 8 * MB);
  u16*   xc1T  = (u16*)(big + 16 * MB);
  float* Tc    = (float*)(big + 24 * MB);
  float* M0    = (float*)(big + 40 * MB);
  float* M1    = (float*)(big + 56 * MB);
  u16*   Mb    = (u16*)(big + 72 * MB);
  float* Xf[2] = {(float*)(big + 80 * MB),  (float*)(big + 96 * MB)};
  u16*   Xb[2] = {(u16*)(big + 112 * MB),   (u16*)(big + 120 * MB)};
  float* T1    = (float*)(big + 128 * MB);
  u16*   T1b   = (u16*)(big + 144 * MB);
  float* U     = (float*)(big + 152 * MB);
  u16*   Qb    = (u16*)(big + 168 * MB);          // 64 MB bf16 queries
  float* raw   = (float*)(big + 232 * MB);

  (void)hipMemsetAsync(tr, 0, 2 * sizeof(float), stream);
  (void)hipMemsetAsync(sc, 0, 2 * sizeof(float), stream);
  (void)hipMemsetAsync(raw, 0, (size_t)NQ * NCLS * sizeof(float), stream);

  k_count<<<1, 256, 0, stream>>>(labels, counts);
  k_colsum<<<DIM / 256, 256, 0, stream>>>(X, labels, counts, mean0, mean1, gmean);
  k_packT<<<dim3(SSUP / 64, DIM / 64), 256, 0, stream>>>(X, labels, mean0, mean1,
                                                         gmean, xtT, xc0T, xc1T);
  k_cvt<<<(NQ * DIM) / 256, 256, 0, stream>>>(Q, Qb);

  dim3 g(DIM / BN, DIM / BM);
  // covariances: NT over [D,S] packs, K = S
  k_gemm_nt<false><<<g, 256, 0, stream>>>(xtT,  xtT,  Tc, SSUP, SSUP, SSUP, DIM,
                                          DIM, DIM, nullptr, nullptr, nullptr, 0);
  k_gemm_nt<false><<<g, 256, 0, stream>>>(xc0T, xc0T, M0, SSUP, SSUP, SSUP, DIM,
                                          DIM, DIM, nullptr, nullptr, nullptr, 0);
  k_gemm_nt<false><<<g, 256, 0, stream>>>(xc1T, xc1T, M1, SSUP, SSUP, SSUP, DIM,
                                          DIM, DIM, nullptr, nullptr, nullptr, 0);
  k_combine<<<(DIM * DIM) / 256, 256, 0, stream>>>(M0, Tc, counts, tr, 0);
  k_combine<<<(DIM * DIM) / 256, 256, 0, stream>>>(M1, Tc, counts, tr, 1);

  for (int c = 0; c < NCLS; ++c) {
    float* Mc = c ? M1 : M0;
    const float* meanc = c ? mean1 : mean0;
    k_cvt<<<(DIM * DIM) / 256, 256, 0, stream>>>(Mc, Mb);
    k_ns_init<<<(DIM * DIM) / 256, 256, 0, stream>>>(tr, Xf[c], Xb[c], c);
    for (int it = 0; it < NS_ITERS; ++it) {
      // T1 = M * X  (X symmetric => NT(Mb, Xb))
      k_gemm_nt<false><<<g, 256, 0, stream>>>(Mb, Xb[c], T1, DIM, DIM, DIM, DIM,
                                              DIM, DIM, nullptr, nullptr, nullptr, 0);
      k_cvt<<<(DIM * DIM) / 256, 256, 0, stream>>>(T1, T1b);
      // U = T1 * X^T = M*X*X (all iterates commute with M); X = 2X - U
      k_gemm_nt<false><<<g, 256, 0, stream>>>(T1b, Xb[c], U, DIM, DIM, DIM, DIM,
                                              DIM, DIM, nullptr, nullptr, nullptr, 0);
      k_update<<<(DIM * DIM) / 256, 256, 0, stream>>>(Xf[c], U, Xb[c]);
    }
    k_sc<<<DIM, 256, 0, stream>>>(Xf[c], meanc, sc, c);
    dim3 gq(DIM / BN, NQ / BM);
    // fused query GEMM: Y = Q * P^T (P symmetric), epilogue reduces over d
    k_gemm_nt<true><<<gq, 256, 0, stream>>>(Qb, Xb[c], nullptr, DIM, DIM, DIM, DIM,
                                            NQ, DIM, Q, meanc, raw, c);
  }
  k_final<<<(NQ * NCLS) / 256, 256, 0, stream>>>(raw, sc, out);
}
